// SparseSpatialEncoder_34583076667421
// MI455X (gfx1250) — matrix operand order repaired and checked by reading, not compiled, not run on hardware
//
#include <hip/hip_runtime.h>
#include <hip/hip_bf16.h>
#include <stdint.h>

// ---------------------------------------------------------------------------
// CDNA5 (gfx1250) sparse U-Net encoder, LDS-free implicit GEMM on
// v_wmma_f32_16x16x32_bf16. Weights pre-swizzled to per-lane B-fragment
// layout (L2-resident); A fragments read directly from global with a
// zero-page select instead of exec-mask predication. bf16 activations, f32
// accumulation, fused BN+ReLU+mask+skip epilogue with a wave-local LDS
// micro-transpose so all bf16 activation stores are coalesced b128.
// ---------------------------------------------------------------------------

typedef __attribute__((ext_vector_type(16))) __bf16 v16bf;
typedef __attribute__((ext_vector_type(8)))  float  v8f;

#define BN_EPS 0.001f

// ------------------------------ utility kernels ----------------------------

__global__ void fill_zero_u32(uint32_t* p, int n) {
  int i = blockIdx.x * blockDim.x + threadIdx.x;
  if (i < n) p[i] = 0u;
}

// feats[n] = voxels[n].sum(points) / nump[n]; scatter into dense bf16 grid
// (channel stride 16, channels 4..15 pre-zeroed) and set occupancy mask m1.
__global__ void scatter_kernel(const float* __restrict__ voxels,
                               const int* __restrict__ coords,
                               const float* __restrict__ nump,
                               __bf16* __restrict__ dense,  // [2,32,128,128,16]
                               float* __restrict__ m1,      // [2,32,128,128]
                               int N) {
  int i = blockIdx.x * blockDim.x + threadIdx.x;
  if (i >= N) return;
  const int P = 5, CIN = 4;
  int b = coords[i * 4 + 0], z = coords[i * 4 + 1];
  int y = coords[i * 4 + 2], x = coords[i * 4 + 3];
  int v = (((b * 32 + z) * 128 + y) * 128 + x);
  float inv = 1.0f / nump[i];
  float f[CIN] = {0.f, 0.f, 0.f, 0.f};
  for (int pp = 0; pp < P; ++pp)
    for (int c = 0; c < CIN; ++c) f[c] += voxels[(i * P + pp) * CIN + c];
  for (int c = 0; c < CIN; ++c) dense[v * 16 + c] = (__bf16)(f[c] * inv);
  m1[v] = 1.0f;
}

// reduce_window(3^3, stride 2, pad 1) > 0
__global__ void mask_down_kernel(const float* __restrict__ mi, float* __restrict__ mo,
                                 int B, int Di, int Hi, int Wi, int Do, int Ho, int Wo) {
  int v = blockIdx.x * blockDim.x + threadIdx.x;
  int NV = B * Do * Ho * Wo;
  if (v >= NV) return;
  int t = v;
  int x = t % Wo; t /= Wo;
  int y = t % Ho; t /= Ho;
  int z = t % Do; int b = t / Do;
  float s = 0.f;
  for (int dz = -1; dz <= 1; ++dz)
    for (int dy = -1; dy <= 1; ++dy)
      for (int dx = -1; dx <= 1; ++dx) {
        int iz = 2 * z + dz, iy = 2 * y + dy, ix = 2 * x + dx;
        if (iz >= 0 && iz < Di && iy >= 0 && iy < Hi && ix >= 0 && ix < Wi)
          s += mi[((b * Di + iz) * Hi + iy) * Wi + ix];
      }
  mo[v] = (s > 0.f) ? 1.0f : 0.0f;
}

// f32 weights (kd,kh,kw,ci,co) -> bf16 in per-lane WMMA B-fragment layout:
//   wb[(((tap*KT + kt)*ntiles + nt)*32 + lane)*16 + e]
//     = w[tap][kt*32 + (lane>>4)*16 + e][nt*16 + (lane&15)]   (0 if k>=Ci)
// so each lane's B fragment is 32 contiguous bytes (2 x b128 loads).
__global__ void wprep_kernel(const float* __restrict__ w, __bf16* __restrict__ wb,
                             int taps, int Ci, int Co, int Kpad) {
  int i = blockIdx.x * blockDim.x + threadIdx.x;
  int n = taps * Kpad * Co;
  if (i >= n) return;
  const int ntiles = Co >> 4, ktiles = Kpad >> 5;
  int e = i & 15;
  int lane = (i >> 4) & 31;
  int t = i >> 9;  // /512 fragments
  int nt = t % ntiles; t /= ntiles;
  int kt = t % ktiles;
  int tap = t / ktiles;
  int k = kt * 32 + (lane >> 4) * 16 + e;
  int co = nt * 16 + (lane & 15);
  float val = (k < Ci) ? w[(tap * Ci + k) * Co + co] : 0.0f;
  wb[i] = (__bf16)val;
}

// Fold BN: scale = g*rsqrt(v+eps); bias = b - m*scale
__global__ void bnprep_kernel(const float* g, const float* b, const float* m,
                              const float* v, float* scale, float* bias, int Co) {
  int i = blockIdx.x * blockDim.x + threadIdx.x;
  if (i >= Co) return;
  float sc = g[i] * rsqrtf(v[i] + BN_EPS);
  scale[i] = sc;
  bias[i] = b[i] - m[i] * sc;
}

// ------------------------------ conv kernel --------------------------------

struct ConvParams {
  const __bf16* in;     // [2,Di,Hi,Wi,Csi]
  const __bf16* wgt;    // B-fragment swizzled weights
  const float* scale;   // [Co]
  const float* bias;    // [Co]
  const float* mask;    // [2,Do,Ho,Wo]
  const __bf16* skip;   // optional [2,Do,Ho,Wo,Cso]
  const __bf16* zpad;   // >=32 zero bf16 (zero-page for invalid taps)
  __bf16* out;          // [2,Do,Ho,Wo,Cso]
  float* out_f32;       // optional final f32 output
  int Di, Hi, Wi, Do, Ho, Wo;
  int Csi, Co, Cso, stride;
};

// 256 threads = 8 independent waves; wave w owns M-tile (blockIdx.x*8+w) of
// 16 output voxels x NT 16-channel N-tiles at blockIdx.y*NT. MODE: 0 = 3x3x3
// conv (stride 1/2, pad 1), 1 = 1x1x1 (pad 0), 2 = transposed (lhs_dil 2,
// pad(1,2)). Per (tap, k-step): the lane loads its A fragment as two
// unconditional b128 loads whose addresses select the zero page when the tap
// is invalid (no exec churn), and each B fragment as two b128 loads from the
// swizzled weight buffer. GEMM main loop uses no LDS and no barriers; EXEC
// is all-ones at every WMMA. The bf16 epilogue routes the D tile through a
// 512B wave-local LDS transpose (in-order DS ops, no barrier) so global
// stores (and skip loads) are coalesced b128.
// All grids launched with NVo % 128 == 0, so no residue guards anywhere.
template <int KT, int NT, int MODE>
__global__ __launch_bounds__(256) void conv3d_wmma_kernel(ConvParams p) {
  __shared__ __align__(16) __bf16 sT[8][16][16];  // per-wave D-tile transpose

  const int lane = threadIdx.x & 31;
  const int wave = threadIdx.x >> 5;
  const int khalf = lane >> 4;   // K half of the A/B fragments
  const int mrow = lane & 15;    // A-matrix row M / D-matrix column N
  const int mtile = blockIdx.x * 8 + wave;
  const int ntiles = p.Co >> 4;
  const int ntbase = blockIdx.y * NT;

  // This lane's A-row voxel (rows duplicated across the two half-waves).
  const int vR = mtile * 16 + mrow;
  int t = vR;
  const int ox = t % p.Wo; t /= p.Wo;
  const int oy = t % p.Ho; t /= p.Ho;
  const int oz = t % p.Do;
  const int ob = t / p.Do;

  const v8f zero8 = {0.f, 0.f, 0.f, 0.f, 0.f, 0.f, 0.f, 0.f};
  v8f acc[NT];
#pragma unroll
  for (int j = 0; j < NT; ++j) acc[j] = zero8;

  const int fragstride = KT * ntiles * 512;  // bf16 elements per tap

  // One GEMM k-sweep for a single tap: A from rowp (or zero page), B from
  // swizzled weights, NT WMMAs reusing the A fragment.
  auto step = [&](const __bf16* rowp, bool ok, const __bf16* wtap) {
#pragma unroll
    for (int kt = 0; kt < KT; ++kt) {
      const int cbase = kt * 32 + khalf * 16;
      const __bf16* s0 = (ok && (cbase + 8) <= p.Csi) ? (rowp + cbase) : p.zpad;
      const __bf16* s1 = (ok && (cbase + 16) <= p.Csi) ? (rowp + cbase + 8) : p.zpad;
      union { uint4 q[2]; v16bf v; } ua;
      ua.q[0] = *(const uint4*)s0;
      ua.q[1] = *(const uint4*)s1;
#pragma unroll
      for (int j = 0; j < NT; ++j) {
        union { uint4 q[2]; v16bf v; } ub;
        const __bf16* wp = wtap + ((kt * ntiles + (ntbase + j)) * 32 + lane) * 16;
        ub.q[0] = ((const uint4*)wp)[0];
        ub.q[1] = ((const uint4*)wp)[1];
        acc[j] = __builtin_amdgcn_wmma_f32_16x16x32_bf16(
            false, ua.v, false, ub.v, (short)0, acc[j], false, false);
      }
    }
  };

  if (MODE == 1) {
    step(p.in + vR * p.Csi, true, p.wgt);
  } else {
    const __bf16* wtap = p.wgt;
    const int ozs = oz * p.stride, oys = oy * p.stride, oxs = ox * p.stride;
    for (int kd = 0; kd < 3; ++kd) {
      int iz; bool okz;
      if (MODE == 2) {
        int nd = oz + kd - 1; okz = ((nd & 1) == 0); iz = nd >> 1;
      } else {
        iz = ozs + kd - 1; okz = true;
      }
      okz = okz && iz >= 0 && iz < p.Di;
      for (int kh = 0; kh < 3; ++kh) {
        int iy; bool oky;
        if (MODE == 2) {
          int nh = oy + kh - 1; oky = ((nh & 1) == 0); iy = nh >> 1;
        } else {
          iy = oys + kh - 1; oky = true;
        }
        oky = okz && oky && iy >= 0 && iy < p.Hi;
        const __bf16* plane = p.in + ((ob * p.Di + iz) * p.Hi + iy) * p.Wi * p.Csi;
        for (int kw = 0; kw < 3; ++kw) {
          int ix; bool okx;
          if (MODE == 2) {
            int nw = ox + kw - 1; okx = ((nw & 1) == 0); ix = nw >> 1;
          } else {
            ix = oxs + kw - 1; okx = true;
          }
          okx = oky && okx && ix >= 0 && ix < p.Wi;
          step(plane + ix * p.Csi, okx, wtap);
          wtap += fragstride;
        }
      }
    }
  }

  // ---- fused epilogue: BN + ReLU + mask (+ skip), coalesced stores ----
  if (p.out_f32) {
    // Final f32 layer: keep the direct (f32) store path.
#pragma unroll
    for (int j = 0; j < NT; ++j) {
      const int col = (ntbase + j) * 16 + mrow;
      const float sc = p.scale[col], bi = p.bias[col];
#pragma unroll
      for (int r = 0; r < 8; ++r) {
        const int mm = khalf * 8 + r;
        const int v = mtile * 16 + mm;
        float val = acc[j][r] * sc + bi;
        val = val > 0.f ? val : 0.f;
        val *= p.mask[v];
        if (p.skip) val += (float)p.skip[v * p.Cso + col];
        p.out_f32[v * p.Cso + col] = val;
      }
    }
  } else {
#pragma unroll
    for (int j = 0; j < NT; ++j) {
      const int col = (ntbase + j) * 16 + mrow;  // D layout: N = lane&15
      const float sc = p.scale[col], bi = p.bias[col];
#pragma unroll
      for (int r = 0; r < 8; ++r) {
        const int mm = khalf * 8 + r;  // D layout: VGPR r holds M = r + 8*lane[4]
        const int v = mtile * 16 + mm;
        float val = acc[j][r] * sc + bi;
        val = val > 0.f ? val : 0.f;
        val *= p.mask[v];
        sT[wave][mm][mrow] = (__bf16)val;  // deposit column into wave-local tile
      }
      // Wave-synchronous transpose read: lane L -> row L>>1, half L&1.
      // DS ops from one wave are in-order; later writes (next j) cannot pass
      // these reads, so no barrier is required.
      const int row = lane >> 1, h = lane & 1;
      const int vrow = mtile * 16 + row;
      const int col0 = (ntbase + j) * 16 + h * 8;
      union { uint4 q; __bf16 e[8]; } dv;
      dv.q = *(const uint4*)&sT[wave][row][h * 8];
      if (p.skip) {
        union { uint4 q; __bf16 e[8]; } sv;
        sv.q = *(const uint4*)(p.skip + vrow * p.Cso + col0);
#pragma unroll
        for (int e = 0; e < 8; ++e)
          dv.e[e] = (__bf16)((float)dv.e[e] + (float)sv.e[e]);
      }
      *(uint4*)(p.out + vrow * p.Cso + col0) = dv.q;  // coalesced b128 store
    }
  }
}

// ------------------------------ host orchestration -------------------------

extern "C" void kernel_launch(void* const* d_in, const int* in_sizes, int n_in,
                              void* d_out, int out_size, void* d_ws, size_t ws_size,
                              hipStream_t stream) {
  (void)n_in; (void)out_size; (void)ws_size;
  const float* voxels = (const float*)d_in[0];
  const int* coords = (const int*)d_in[1];
  const float* nump = (const float*)d_in[2];

  struct LayerDef { int k, ci, co; };
  static const LayerDef LD[24] = {
      {3, 4, 16},  {3, 16, 16},                             // e1a e1b
      {3, 16, 32}, {3, 32, 32}, {3, 32, 32},                // e2a e2b e2c
      {3, 32, 64}, {3, 64, 64}, {3, 64, 64},                // e3a e3b e3c
      {3, 64, 128},{3, 128, 128},{3, 128, 128},             // e4a e4b e4c
      {1, 128, 128},                                        // c4
      {3, 128, 64},{3, 64, 64}, {3, 64, 64}, {1, 64, 64},   // d3a d3b d3c c3
      {3, 64, 32}, {3, 32, 32}, {3, 32, 32}, {1, 32, 32},   // d2a d2b d2c c2
      {3, 32, 16}, {3, 16, 16}, {3, 16, 16}, {1, 16, 16}};  // d1a d1b d1c c1

  const int NV1 = 2 * 32 * 128 * 128;
  const int NV2 = 2 * 16 * 64 * 64;
  const int NV3 = 2 * 8 * 32 * 32;
  const int NV4 = 2 * 4 * 16 * 16;

  // --- carve workspace ---
  char* ws = (char*)d_ws;
  size_t off = 0;
  auto alloc = [&](size_t bytes) -> char* {
    char* p = ws + off;
    off += (bytes + 255) & ~(size_t)255;
    return p;
  };
  __bf16* ZP = (__bf16*)alloc(256);                   // zero page
  __bf16* A  = (__bf16*)alloc((size_t)NV1 * 16 * 2);  // dense input; reused as Q1
  __bf16* X1 = (__bf16*)alloc((size_t)NV1 * 16 * 2);
  __bf16* P1 = (__bf16*)alloc((size_t)NV1 * 16 * 2);
  __bf16* Q1 = A;
  __bf16* X2 = (__bf16*)alloc((size_t)NV2 * 32 * 2);
  __bf16* P2 = (__bf16*)alloc((size_t)NV2 * 32 * 2);
  __bf16* Q2 = (__bf16*)alloc((size_t)NV2 * 32 * 2);
  __bf16* X3 = (__bf16*)alloc((size_t)NV3 * 64 * 2);
  __bf16* P3 = (__bf16*)alloc((size_t)NV3 * 64 * 2);
  __bf16* Q3 = (__bf16*)alloc((size_t)NV3 * 64 * 2);
  __bf16* X4 = (__bf16*)alloc((size_t)NV4 * 128 * 2);
  __bf16* P4 = (__bf16*)alloc((size_t)NV4 * 128 * 2);
  __bf16* Q4 = (__bf16*)alloc((size_t)NV4 * 128 * 2);
  float* M1 = (float*)alloc((size_t)NV1 * 4);
  float* M2 = (float*)alloc((size_t)NV2 * 4);
  float* M3 = (float*)alloc((size_t)NV3 * 4);
  float* M4 = (float*)alloc((size_t)NV4 * 4);
  __bf16* WB[24]; float* SC[24]; float* BI[24];
  for (int i = 0; i < 24; ++i) {
    int taps = (LD[i].k == 3) ? 27 : 1;
    int kpad = (LD[i].ci + 31) & ~31;
    WB[i] = (__bf16*)alloc((size_t)taps * kpad * LD[i].co * 2);
    SC[i] = (float*)alloc((size_t)LD[i].co * 4);
    BI[i] = (float*)alloc((size_t)LD[i].co * 4);
  }

  // --- input prep ---
  fill_zero_u32<<<dim3(1), dim3(64), 0, stream>>>((uint32_t*)ZP, 64);
  {
    int nd = NV1 * 16 / 2;  // dense bf16 elements as u32 words
    fill_zero_u32<<<dim3((nd + 255) / 256), dim3(256), 0, stream>>>((uint32_t*)A, nd);
    fill_zero_u32<<<dim3((NV1 + 255) / 256), dim3(256), 0, stream>>>((uint32_t*)M1, NV1);
  }
  const int Nvox = in_sizes[2];
  scatter_kernel<<<dim3((Nvox + 255) / 256), dim3(256), 0, stream>>>(
      voxels, coords, nump, A, M1, Nvox);
  mask_down_kernel<<<dim3((NV2 + 255) / 256), dim3(256), 0, stream>>>(
      M1, M2, 2, 32, 128, 128, 16, 64, 64);
  mask_down_kernel<<<dim3((NV3 + 255) / 256), dim3(256), 0, stream>>>(
      M2, M3, 2, 16, 64, 64, 8, 32, 32);
  mask_down_kernel<<<dim3((NV4 + 255) / 256), dim3(256), 0, stream>>>(
      M3, M4, 2, 8, 32, 32, 4, 16, 16);

  for (int i = 0; i < 24; ++i) {
    int taps = (LD[i].k == 3) ? 27 : 1;
    int kpad = (LD[i].ci + 31) & ~31;
    const float* w = (const float*)d_in[3 + 5 * i + 0];
    const float* g = (const float*)d_in[3 + 5 * i + 1];
    const float* b = (const float*)d_in[3 + 5 * i + 2];
    const float* m = (const float*)d_in[3 + 5 * i + 3];
    const float* v = (const float*)d_in[3 + 5 * i + 4];
    int n = taps * kpad * LD[i].co;
    wprep_kernel<<<dim3((n + 255) / 256), dim3(256), 0, stream>>>(
        w, WB[i], taps, LD[i].ci, LD[i].co, kpad);
    bnprep_kernel<<<dim3((LD[i].co + 63) / 64), dim3(64), 0, stream>>>(
        g, b, m, v, SC[i], BI[i], LD[i].co);
  }

  // --- conv launcher with (KT, NT, MODE) template dispatch ---
  struct Dims { int D, H, W; };
  const Dims DL[5] = {{0, 0, 0}, {32, 128, 128}, {16, 64, 64}, {8, 32, 32}, {4, 16, 16}};
  float* const Mlev[5] = {nullptr, M1, M2, M3, M4};
  auto conv = [&](int li, const __bf16* in, int lin, int lout, int Csi, int Cso,
                  const __bf16* skip, __bf16* out, float* outf, int stride, int mode) {
    ConvParams p{};
    p.in = in; p.wgt = WB[li]; p.scale = SC[li]; p.bias = BI[li];
    p.mask = Mlev[lout]; p.skip = skip; p.zpad = ZP; p.out = out; p.out_f32 = outf;
    p.Di = DL[lin].D;  p.Hi = DL[lin].H;  p.Wi = DL[lin].W;
    p.Do = DL[lout].D; p.Ho = DL[lout].H; p.Wo = DL[lout].W;
    p.Csi = Csi; p.Co = LD[li].co; p.Cso = Cso; p.stride = stride;
    const int KT = ((LD[li].ci + 31) & ~31) >> 5;  // k-steps of 32
    const int NT = (p.Co >= 32) ? 2 : 1;
    const int NVo = 2 * p.Do * p.Ho * p.Wo;
    dim3 grid((unsigned)(NVo / 128), (unsigned)(p.Co / 16 / NT), 1);
    dim3 blk(256);
#define LC(K, N, M) conv3d_wmma_kernel<K, N, M><<<grid, blk, 0, stream>>>(p)
    if (mode == 0) {
      if      (KT == 1 && NT == 1) LC(1, 1, 0);
      else if (KT == 1 && NT == 2) LC(1, 2, 0);
      else if (KT == 2 && NT == 2) LC(2, 2, 0);
      else                         LC(4, 2, 0);
    } else if (mode == 1) {
      if      (KT == 1 && NT == 1) LC(1, 1, 1);
      else if (KT == 1 && NT == 2) LC(1, 2, 1);
      else if (KT == 2 && NT == 2) LC(2, 2, 1);
      else                         LC(4, 2, 1);
    } else {
      if      (KT == 1 && NT == 1) LC(1, 1, 2);
      else if (KT == 2 && NT == 2) LC(2, 2, 2);
      else                         LC(4, 2, 2);
    }
#undef LC
  };

  // --- encoder ---
  conv(0,  A,  1, 1, 16, 16,  nullptr, P1, nullptr, 1, 0);  // e1a
  conv(1,  P1, 1, 1, 16, 16,  nullptr, X1, nullptr, 1, 0);  // e1b
  conv(2,  X1, 1, 2, 16, 32,  nullptr, P2, nullptr, 2, 0);  // e2a (stride 2)
  conv(3,  P2, 2, 2, 32, 32,  nullptr, Q2, nullptr, 1, 0);  // e2b
  conv(4,  Q2, 2, 2, 32, 32,  nullptr, X2, nullptr, 1, 0);  // e2c
  conv(5,  X2, 2, 3, 32, 64,  nullptr, P3, nullptr, 2, 0);  // e3a
  conv(6,  P3, 3, 3, 64, 64,  nullptr, Q3, nullptr, 1, 0);  // e3b
  conv(7,  Q3, 3, 3, 64, 64,  nullptr, X3, nullptr, 1, 0);  // e3c
  conv(8,  X3, 3, 4, 64, 128, nullptr, P4, nullptr, 2, 0);  // e4a
  conv(9,  P4, 4, 4, 128, 128, nullptr, Q4, nullptr, 1, 0); // e4b
  conv(10, Q4, 4, 4, 128, 128, nullptr, X4, nullptr, 1, 0); // e4c
  conv(11, X4, 4, 4, 128, 128, nullptr, P4, nullptr, 1, 1); // c4 -> f4
  // --- decoder level 3 ---
  conv(12, P4, 4, 3, 128, 64, nullptr, P3, nullptr, 1, 2);  // d3a (deconv)
  conv(13, P3, 3, 3, 64, 64,  nullptr, Q3, nullptr, 1, 0);  // d3b
  conv(14, Q3, 3, 3, 64, 64,  nullptr, P3, nullptr, 1, 0);  // d3c -> y3
  conv(15, X3, 3, 3, 64, 64,  P3,      Q3, nullptr, 1, 1);  // c3 + y3 -> f3
  // --- decoder level 2 ---
  conv(16, Q3, 3, 2, 64, 32,  nullptr, P2, nullptr, 1, 2);  // d2a (deconv)
  conv(17, P2, 2, 2, 32, 32,  nullptr, Q2, nullptr, 1, 0);  // d2b
  conv(18, Q2, 2, 2, 32, 32,  nullptr, P2, nullptr, 1, 0);  // d2c -> y2
  conv(19, X2, 2, 2, 32, 32,  P2,      Q2, nullptr, 1, 1);  // c2 + y2 -> f2
  // --- decoder level 1 ---
  conv(20, Q2, 2, 1, 32, 16,  nullptr, P1, nullptr, 1, 2);  // d1a (deconv)
  conv(21, P1, 1, 1, 16, 16,  nullptr, Q1, nullptr, 1, 0);  // d1b
  conv(22, Q1, 1, 1, 16, 16,  nullptr, P1, nullptr, 1, 0);  // d1c -> y1
  conv(23, X1, 1, 1, 16, 16,  P1, nullptr, (float*)d_out, 1, 1);  // c1 + y1 -> f1
}
